// MambaClassifier_80564996539263
// MI455X (gfx1250) — compile-verified, matrix operand order checked by reading, not run on hardware
//
#include <hip/hip_runtime.h>
#include <hip/hip_bf16.h>

// ---------------------------------------------------------------------------
// Mamba classifier forward for MI455X (gfx1250, wave32, WMMA).
// Shapes: B=4 T=512 R=64 D_EMB=512 D_MODEL=256 D_IN=512 D_STATE=16
//         D_CONV=4 DT_RANK=16 N_LAYERS=4
// ---------------------------------------------------------------------------

typedef __attribute__((ext_vector_type(16))) _Float16 v16h;
typedef __attribute__((ext_vector_type(4)))  _Float16 v4h;
typedef __attribute__((ext_vector_type(8)))  float    v8f;

#define Bb      4
#define Tt      512
#define Rr      64
#define DEMB    512
#define DMODEL  256
#define DIN     512
#define DSTATE  16
#define DTRANK  16
#define NROWS   (Bb * Tt)          // 2048 flattened (b,t) rows

__device__ __forceinline__ float silu_f(float x) {
    return x / (1.0f + __expf(-x));
}

// ---------------------------------------------------------------------------
// 1) Frame pooling: masked mean over R, layernorm over D_EMB -> xpool[2048][512]
//    esmif is streamed exactly once; each thread owns 2 consecutive columns so
//    loads coalesce/merge, with non-temporal hint (no reuse).
// ---------------------------------------------------------------------------
__global__ void pool_ln_kernel(const float* __restrict__ esmif,
                               const unsigned char* __restrict__ rmask,
                               const float* __restrict__ g,
                               const float* __restrict__ bta,
                               float* __restrict__ out) {
    const int frame = blockIdx.x;            // 0..2047
    const int tid   = threadIdx.x;           // 256 threads, 2 adjacent cols each
    const int col   = tid * 2;
    __shared__ float red[256];
    __shared__ float s_mean, s_rstd, s_n;

    const float* base = esmif + (size_t)frame * Rr * DEMB;
    const unsigned char* mk = rmask + (size_t)frame * Rr;

    float acc0 = 0.0f, acc1 = 0.0f, cnt = 0.0f;
    for (int r = 0; r < Rr; ++r) {
        float mv = mk[r] ? 1.0f : 0.0f;
        if (tid == 0) cnt += mv;
        acc0 += mv * __builtin_nontemporal_load(&base[r * DEMB + col]);
        acc1 += mv * __builtin_nontemporal_load(&base[r * DEMB + col + 1]);
    }
    if (tid == 0) s_n = fmaxf(cnt, 1.0f);
    __syncthreads();
    const float inv_n = 1.0f / s_n;
    float x0 = acc0 * inv_n, x1 = acc1 * inv_n;

    red[tid] = x0 + x1; __syncthreads();
    for (int s = 128; s > 0; s >>= 1) { if (tid < s) red[tid] += red[tid + s]; __syncthreads(); }
    if (tid == 0) s_mean = red[0] * (1.0f / (float)DEMB);
    __syncthreads();
    const float mu = s_mean;
    float d0 = x0 - mu, d1 = x1 - mu;
    red[tid] = d0 * d0 + d1 * d1; __syncthreads();
    for (int s = 128; s > 0; s >>= 1) { if (tid < s) red[tid] += red[tid + s]; __syncthreads(); }
    if (tid == 0) s_rstd = rsqrtf(red[0] * (1.0f / (float)DEMB) + 1e-5f);
    __syncthreads();
    const float rs = s_rstd;
    out[(size_t)frame * DEMB + col]     = d0 * rs * g[col]     + bta[col];
    out[(size_t)frame * DEMB + col + 1] = d1 * rs * g[col + 1] + bta[col + 1];
}

// ---------------------------------------------------------------------------
// 2) WMMA GEMM: C[M,N] = A[M,K] @ B[K,N] (+bias[n]) (+resid) (*rowmask)
//    Block = 256 thr = 8 waves; block tile 128(M) x 32(N); each wave owns a
//    16x16 M-tile and TWO N-tiles (A-fragment reuse -> 2 v_wmma per K-step).
//    f32 tiles staged to LDS as f16 via float4 loads + packed 8B LDS stores.
//    M % 128 == 0 and K % 32 == 0 always hold here.
//    GUARD_N is a compile-time flag: only the N=48 (x_proj) GEMM needs bounds
//    checks; all other GEMMs get a branch-free inner loop.
// ---------------------------------------------------------------------------
template <bool GUARD_N>
__global__ void gemm_wmma_kernel(const float* __restrict__ A,
                                 const float* __restrict__ B,
                                 const float* __restrict__ bias,
                                 const float* __restrict__ resid,
                                 const unsigned char* __restrict__ rowmask,
                                 float* __restrict__ C,
                                 int M, int N, int K) {
    // rows padded to 40 f16 (80 B): 8-byte aligned groups, conflict-free gather
    __shared__ _Float16 As[128][40];   // [m][k]
    __shared__ _Float16 Bs[32][40];    // transposed: [n][k]

    const int m0     = blockIdx.x * 128;
    const int n0     = blockIdx.y * 32;
    const int tid    = threadIdx.x;
    const int lane   = tid & 31;
    const int wave   = tid >> 5;       // 0..7
    const int laneLo = lane & 15;
    const int laneHi = lane >> 4;

    v8f acc0 = {}, acc1 = {};

    for (int k0 = 0; k0 < K; k0 += 32) {
        // --- A tile: 128x32 f32 -> f16. 1024 float4 chunks, 4 per thread. ---
        #pragma unroll
        for (int e = 0; e < 4; ++e) {
            int chunk = e * 256 + tid;          // 0..1023
            int r  = chunk >> 3;                // row 0..127
            int c4 = (chunk & 7) << 2;          // col 0,4,...,28
            float4 t = *(const float4*)&A[(size_t)(m0 + r) * K + (k0 + c4)];
            v4h p;
            p[0] = (_Float16)t.x; p[1] = (_Float16)t.y;
            p[2] = (_Float16)t.z; p[3] = (_Float16)t.w;
            *(v4h*)&As[r][c4] = p;              // single 8B ds_store
        }
        // --- B tile: 32(k) x 32(n) -> transposed. One float4 per thread. ---
        {
            int kk = tid >> 3;                  // 0..31
            int n4 = (tid & 7) << 2;            // 0,4,...,28
            int gn = n0 + n4;
            float b0, b1, b2, b3;
            if (!GUARD_N || (gn + 3 < N)) {
                float4 t = *(const float4*)&B[(size_t)(k0 + kk) * N + gn];
                b0 = t.x; b1 = t.y; b2 = t.z; b3 = t.w;
            } else {
                b0 = (gn + 0 < N) ? B[(size_t)(k0 + kk) * N + gn + 0] : 0.0f;
                b1 = (gn + 1 < N) ? B[(size_t)(k0 + kk) * N + gn + 1] : 0.0f;
                b2 = (gn + 2 < N) ? B[(size_t)(k0 + kk) * N + gn + 2] : 0.0f;
                b3 = (gn + 3 < N) ? B[(size_t)(k0 + kk) * N + gn + 3] : 0.0f;
            }
            Bs[n4 + 0][kk] = (_Float16)b0;
            Bs[n4 + 1][kk] = (_Float16)b1;
            Bs[n4 + 2][kk] = (_Float16)b2;
            Bs[n4 + 3][kk] = (_Float16)b3;
        }
        if (k0 + 32 < K)  // speculative prefetch of next A tile
            __builtin_prefetch(&A[(size_t)(m0 + (tid >> 1)) * K + (k0 + 32)], 0, 0);
        __syncthreads();

        // Fragments per ISA 16-bit 16x32 A / 32x16 B layouts:
        // lanes 0-15 hold K 0..7 & 16..23, lanes 16-31 hold K 8..15 & 24..31.
        v16h af, bf0, bf1;
        #pragma unroll
        for (int e = 0; e < 16; ++e) {
            int kk = ((e < 8) ? e : (e + 8)) + laneHi * 8;
            af[e]  = As[wave * 16 + laneLo][kk];
            bf0[e] = Bs[laneLo][kk];
            bf1[e] = Bs[16 + laneLo][kk];
        }
        acc0 = __builtin_amdgcn_wmma_f32_16x16x32_f16(
            false, af, false, bf0, (short)0, acc0, false, false);
        acc1 = __builtin_amdgcn_wmma_f32_16x16x32_f16(
            false, af, false, bf1, (short)0, acc1, false, false);
        __syncthreads();
    }

    // C/D layout: VGPR r -> M = r (lanes 0-15) or r+8 (lanes 16-31), N = lane&15
    #pragma unroll
    for (int r = 0; r < 8; ++r) {
        int m = m0 + wave * 16 + laneHi * 8 + r;
        int n = n0 + laneLo;
        if (!GUARD_N || (n < N)) {
            float v = acc0[r];
            if (bias)    v += bias[n];
            if (resid)   v += resid[(size_t)m * N + n];
            if (rowmask) v *= (rowmask[m] ? 0.0f : 1.0f);
            C[(size_t)m * N + n] = v;
        }
        int n2 = n + 16;
        if (!GUARD_N || (n2 < N)) {
            float v = acc1[r];
            if (bias)    v += bias[n2];
            if (resid)   v += resid[(size_t)m * N + n2];
            if (rowmask) v *= (rowmask[m] ? 0.0f : 1.0f);
            C[(size_t)m * N + n2] = v;
        }
    }
}

// ---------------------------------------------------------------------------
// 3) h = frame + pos_emb[frame_idxs]   (in place on h)
// ---------------------------------------------------------------------------
__global__ void add_posemb_kernel(float* __restrict__ h,
                                  const float* __restrict__ pos,
                                  const int* __restrict__ idxs) {
    int i = blockIdx.x * blockDim.x + threadIdx.x;   // 2048*256
    int row = i >> 8, d = i & 255;
    h[i] += pos[(size_t)idxs[row] * DMODEL + d];
}

// ---------------------------------------------------------------------------
// 4) Layernorm over D_MODEL=256 (one block per row)
// ---------------------------------------------------------------------------
__global__ void ln256_kernel(const float* __restrict__ x,
                             const float* __restrict__ g,
                             const float* __restrict__ bta,
                             float* __restrict__ out) {
    const int row = blockIdx.x;
    const int tid = threadIdx.x;   // 256
    __shared__ float red[256];
    __shared__ float s_mean, s_rstd;
    float v = x[(size_t)row * DMODEL + tid];
    red[tid] = v; __syncthreads();
    for (int s = 128; s > 0; s >>= 1) { if (tid < s) red[tid] += red[tid + s]; __syncthreads(); }
    if (tid == 0) s_mean = red[0] * (1.0f / (float)DMODEL);
    __syncthreads();
    float d = v - s_mean;
    red[tid] = d * d; __syncthreads();
    for (int s = 128; s > 0; s >>= 1) { if (tid < s) red[tid] += red[tid + s]; __syncthreads(); }
    if (tid == 0) s_rstd = rsqrtf(red[0] * (1.0f / (float)DMODEL) + 1e-5f);
    __syncthreads();
    out[(size_t)row * DMODEL + tid] = d * s_rstd * g[tid] + bta[tid];
}

// ---------------------------------------------------------------------------
// 5) depthwise causal conv (d_conv=4) + silu : u2 = silu(conv(u)), u = uz[:,:512]
// ---------------------------------------------------------------------------
__global__ void conv_silu_kernel(const float* __restrict__ uz,
                                 const float* __restrict__ cw,
                                 const float* __restrict__ cb,
                                 float* __restrict__ u2) {
    int i = blockIdx.x * blockDim.x + threadIdx.x;   // 2048*512
    int row = i >> 9, c = i & 511;
    int t = row & (Tt - 1);
    float acc = cb[c];
    #pragma unroll
    for (int k = 0; k < 4; ++k) {
        int tp = t - 3 + k;
        if (tp >= 0)
            acc += cw[c * 4 + k] * uz[(size_t)(row - 3 + k) * (2 * DIN) + c];
    }
    u2[i] = silu_f(acc);
}

// ---------------------------------------------------------------------------
// 6) delta = softplus(dt @ W_dt + b_dt), dt = xdbc[:, :16]   (K=16 -> VALU)
// ---------------------------------------------------------------------------
__global__ void dt_softplus_kernel(const float* __restrict__ xdbc,
                                   const float* __restrict__ Wdt,
                                   const float* __restrict__ bdt,
                                   float* __restrict__ delta) {
    int i = blockIdx.x * blockDim.x + threadIdx.x;   // 2048*512
    int row = i >> 9, c = i & 511;
    const float* xr = xdbc + (size_t)row * (DTRANK + 2 * DSTATE);
    float acc = bdt[c];
    #pragma unroll
    for (int j = 0; j < DTRANK; ++j) acc += xr[j] * Wdt[j * DIN + c];
    delta[i] = (acc > 20.0f) ? acc : log1pf(__expf(acc));
}

// ---------------------------------------------------------------------------
// 7) Selective scan: states in registers (16/thread), B/C staged in LDS.
//    grid (2,4): blockIdx.y = batch, blockIdx.x*256+tid = channel.
//    y = (scan + Dp*u) * silu(z)
// ---------------------------------------------------------------------------
__global__ void scan_kernel(const float* __restrict__ delta,
                            const float* __restrict__ u2,
                            const float* __restrict__ uz,
                            const float* __restrict__ xdbc,
                            const float* __restrict__ A_log,
                            const float* __restrict__ Dp,
                            float* __restrict__ y) {
    const int b = blockIdx.y;
    const int c = blockIdx.x * 256 + threadIdx.x;  // 0..511
    __shared__ float Bs[DSTATE], Cs[DSTATE];

    float Ac[DSTATE], hs[DSTATE];
    #pragma unroll
    for (int n = 0; n < DSTATE; ++n) {
        Ac[n] = -__expf(A_log[(size_t)c * DSTATE + n]);
        hs[n] = 0.0f;
    }
    const float dcoef = Dp[c];

    for (int t = 0; t < Tt; ++t) {
        const size_t row = (size_t)b * Tt + t;
        if (threadIdx.x < DSTATE)
            Bs[threadIdx.x] = xdbc[row * (DTRANK + 2 * DSTATE) + DTRANK + threadIdx.x];
        else if (threadIdx.x < 2 * DSTATE)
            Cs[threadIdx.x - DSTATE] =
                xdbc[row * (DTRANK + 2 * DSTATE) + DTRANK + DSTATE + (threadIdx.x - DSTATE)];
        __syncthreads();

        const float dlt = delta[row * DIN + c];
        const float uu  = u2[row * DIN + c];
        float acc = 0.0f;
        #pragma unroll
        for (int n = 0; n < DSTATE; ++n) {
            float dA = __expf(dlt * Ac[n]);
            hs[n] = dA * hs[n] + dlt * Bs[n] * uu;
            acc += hs[n] * Cs[n];
        }
        acc += dcoef * uu;
        const float z = uz[row * (2 * DIN) + DIN + c];
        acc *= silu_f(z);
        y[row * DIN + c] = acc;
        __syncthreads();
    }
}

// ---------------------------------------------------------------------------
// 8) Head: pick h[b, len-1], LN(256), dot head_W + head_b -> out[b]
// ---------------------------------------------------------------------------
__global__ void head_kernel(const float* __restrict__ h,
                            const unsigned char* __restrict__ tmask,
                            const float* __restrict__ g,
                            const float* __restrict__ bta,
                            const float* __restrict__ W,
                            const float* __restrict__ hb,
                            float* __restrict__ out) {
    __shared__ float red[256];
    __shared__ float s_mean, s_rstd;
    const int tid = threadIdx.x;
    for (int b = 0; b < Bb; ++b) {
        int cnt = 0;
        for (int t = tid; t < Tt; t += 256) cnt += tmask[b * Tt + t] ? 0 : 1;
        red[tid] = (float)cnt; __syncthreads();
        for (int s = 128; s > 0; s >>= 1) { if (tid < s) red[tid] += red[tid + s]; __syncthreads(); }
        int len = (int)red[0]; if (len < 1) len = 1;
        __syncthreads();

        const float x = h[((size_t)b * Tt + (len - 1)) * DMODEL + tid];
        red[tid] = x; __syncthreads();
        for (int s = 128; s > 0; s >>= 1) { if (tid < s) red[tid] += red[tid + s]; __syncthreads(); }
        if (tid == 0) s_mean = red[0] * (1.0f / (float)DMODEL);
        __syncthreads();
        const float d = x - s_mean;
        red[tid] = d * d; __syncthreads();
        for (int s = 128; s > 0; s >>= 1) { if (tid < s) red[tid] += red[tid + s]; __syncthreads(); }
        if (tid == 0) s_rstd = rsqrtf(red[0] * (1.0f / (float)DMODEL) + 1e-5f);
        __syncthreads();
        const float xn = d * s_rstd * g[tid] + bta[tid];
        red[tid] = xn * W[tid]; __syncthreads();
        for (int s = 128; s > 0; s >>= 1) { if (tid < s) red[tid] += red[tid + s]; __syncthreads(); }
        if (tid == 0) out[b] = red[0] + hb[0];
        __syncthreads();
    }
}

// ---------------------------------------------------------------------------
// Host orchestration
// ---------------------------------------------------------------------------
extern "C" void kernel_launch(void* const* d_in, const int* in_sizes, int n_in,
                              void* d_out, int out_size, void* d_ws, size_t ws_size,
                              hipStream_t stream) {
    (void)in_sizes; (void)n_in; (void)out_size; (void)ws_size;

    const float* esmif     = (const float*)d_in[0];
    const float* pool_ln_g = (const float*)d_in[1];
    const float* pool_ln_b = (const float*)d_in[2];
    const float* pool_W    = (const float*)d_in[3];
    const float* pool_b    = (const float*)d_in[4];
    const float* pos_emb   = (const float*)d_in[5];
    const float* ln_g      = (const float*)d_in[6];
    const float* ln_b      = (const float*)d_in[7];
    const float* W_in      = (const float*)d_in[8];
    const float* conv_w    = (const float*)d_in[9];
    const float* conv_b    = (const float*)d_in[10];
    const float* W_x       = (const float*)d_in[11];
    const float* W_dt      = (const float*)d_in[12];
    const float* b_dt      = (const float*)d_in[13];
    const float* A_log     = (const float*)d_in[14];
    const float* Dp        = (const float*)d_in[15];
    const float* W_out     = (const float*)d_in[16];
    const float* head_ln_g = (const float*)d_in[17];
    const float* head_ln_b = (const float*)d_in[18];
    const float* head_W    = (const float*)d_in[19];
    const float* head_b    = (const float*)d_in[20];
    const unsigned char* res_mask  = (const unsigned char*)d_in[21];
    const unsigned char* time_mask = (const unsigned char*)d_in[22];
    const int* frame_idxs  = (const int*)d_in[23];

    float* out = (float*)d_out;

    // workspace partition (floats)
    float* ws    = (float*)d_ws;
    float* xpool = ws;                       // 2048*512
    float* h     = xpool + NROWS * DEMB;     // 2048*256
    float* xln   = h     + NROWS * DMODEL;   // 2048*256
    float* uz    = xln   + NROWS * DMODEL;   // 2048*1024
    float* u2    = uz    + NROWS * 2 * DIN;  // 2048*512
    float* xdbc  = u2    + NROWS * DIN;      // 2048*48
    float* delta = xdbc  + NROWS * (DTRANK + 2 * DSTATE); // 2048*512
    float* ybuf  = delta + NROWS * DIN;      // 2048*512

    const int NDBC = DTRANK + 2 * DSTATE;    // 48

    // 1) frame pooling + LN (streams 268 MB of esmif once, NT hint)
    pool_ln_kernel<<<NROWS, 256, 0, stream>>>(esmif, res_mask, pool_ln_g, pool_ln_b, xpool);

    // 2) frame = xpool @ pool_W + pool_b  -> h   (N=256, no guards)
    gemm_wmma_kernel<false><<<dim3(NROWS / 128, DMODEL / 32), 256, 0, stream>>>(
        xpool, pool_W, pool_b, nullptr, nullptr, h, NROWS, DMODEL, DEMB);

    // 3) h += pos_emb[frame_idxs]
    add_posemb_kernel<<<(NROWS * DMODEL) / 256, 256, 0, stream>>>(h, pos_emb, frame_idxs);

    // 4) mamba layers
    for (int l = 0; l < 4; ++l) {
        ln256_kernel<<<NROWS, 256, 0, stream>>>(h, ln_g + l * DMODEL, ln_b + l * DMODEL, xln);

        // in_proj: N=1024, no guards
        gemm_wmma_kernel<false><<<dim3(NROWS / 128, (2 * DIN) / 32), 256, 0, stream>>>(
            xln, W_in + (size_t)l * DMODEL * 2 * DIN, nullptr, nullptr, nullptr,
            uz, NROWS, 2 * DIN, DMODEL);

        conv_silu_kernel<<<(NROWS * DIN) / 256, 256, 0, stream>>>(
            uz, conv_w + (size_t)l * DIN * 4, conv_b + (size_t)l * DIN, u2);

        // x_proj: N=48 -> guarded instantiation
        gemm_wmma_kernel<true><<<dim3(NROWS / 128, (NDBC + 31) / 32), 256, 0, stream>>>(
            u2, W_x + (size_t)l * DIN * NDBC, nullptr, nullptr, nullptr,
            xdbc, NROWS, NDBC, DIN);

        dt_softplus_kernel<<<(NROWS * DIN) / 256, 256, 0, stream>>>(
            xdbc, W_dt + (size_t)l * DTRANK * DIN, b_dt + (size_t)l * DIN, delta);

        scan_kernel<<<dim3(DIN / 256, Bb), 256, 0, stream>>>(
            delta, u2, uz, xdbc, A_log + (size_t)l * DIN * DSTATE, Dp + (size_t)l * DIN, ybuf);

        // out_proj: N=256, no guards; h = (h + y @ W_out) * pad fused
        gemm_wmma_kernel<false><<<dim3(NROWS / 128, DMODEL / 32), 256, 0, stream>>>(
            ybuf, W_out + (size_t)l * DIN * DMODEL, nullptr, h, time_mask,
            h, NROWS, DMODEL, DIN);
    }

    // 5) head
    head_kernel<<<1, 256, 0, stream>>>(h, time_mask, head_ln_g, head_ln_b, head_W, head_b, out);
}
